// SubgraphGCN_76029511074162
// MI455X (gfx1250) — matrix operand order repaired and checked by reading, not compiled, not run on hardware
//
#include <hip/hip_runtime.h>
#include <hip/hip_bf16.h>

typedef __attribute__((ext_vector_type(16))) _Float16 v16h;
typedef __attribute__((ext_vector_type(8)))  _Float16 v8h;
typedef __attribute__((ext_vector_type(8)))  float    v8f;

constexpr int G  = 64;
constexpr int N  = 2048;   // nodes per graph (incl. center)
constexpr int NN = 2047;   // nodes per graph excluding center (x_nc)
constexpr int E  = 16384;
constexpr int GE = G * E;  // 1,048,576 total edges
constexpr int F  = 128;    // feature / hidden / output dims (all 128)
constexpr int NP = 2048;   // padded per-graph node stride in workspace

// ---------------------------------------------------------------------------
// Fragment builders following CDNA5 ISA 7.12.2 layouts (wave32).
// A (16x32 f16): lanes 0-15 => M=lane, K in {base..base+7, base+16..base+23},
// base = (lane>=16 ? 8 : 0).  Two contiguous 8-half segments -> 2x b128 loads.
__device__ inline v16h load_fragA(const _Float16* row128, int kc, int lane) {
  const v8h* rp = (const v8h*)row128;
  int b = kc * 4 + (lane >> 4);      // (kc*32 + (lane>>4)*8) / 8
  v8h lo = rp[b];
  v8h hi = rp[b + 2];                // +16 halves
  return __builtin_shufflevector(lo, hi, 0, 1, 2, 3, 4, 5, 6, 7,
                                 8, 9, 10, 11, 12, 13, 14, 15);
}

// B (32x16 f16) pre-swizzled in global memory: entry = ((w*8+nt)*4+kc)*32+lane,
// 16 contiguous halves (K = kc*32 + (lane>=16?8:0) + (e&7) + (e>>3)*16, N = nt*16 + (lane&15)).
__device__ inline v16h load_fragB(const _Float16* wfrag, int w, int nt, int kc,
                                  int lane) {
  const _Float16* p =
      wfrag + ((size_t)(((w * 8 + nt) * 4 + kc) * 32 + lane)) * 16;
  v8h lo = *(const v8h*)p;
  v8h hi = *(const v8h*)(p + 8);
  return __builtin_shufflevector(lo, hi, 0, 1, 2, 3, 4, 5, 6, 7,
                                 8, 9, 10, 11, 12, 13, 14, 15);
}

// ---------------------------------------------------------------------------
__global__ void k_zero(float* p, int n) {
  int i = blockIdx.x * blockDim.x + threadIdx.x;
  if (i < n) p[i] = 0.f;
}

// degree over dst (x_nc graph: node ids 0..2046 only, by construction)
__global__ void k_deg(const int* __restrict__ ei, float* __restrict__ deg) {
  int e = blockIdx.x * blockDim.x + threadIdx.x;
  if (e >= GE) return;
  int d = ei[GE + e];                      // global dst id = g*N + local
  atomicAdd(deg + (size_t)(d >> 11) * NP + (d & 2047), 1.0f);
}

__global__ void k_dis(const float* __restrict__ deg, float* __restrict__ dis,
                      float* __restrict__ invd) {
  int i = blockIdx.x * blockDim.x + threadIdx.x;
  if (i >= G * NP) return;
  int node = i & (NP - 1);
  if (node < NN) {
    float d = 1.f + deg[i];
    dis[i]  = rsqrtf(d);
    invd[i] = 1.f / d;
  } else {
    dis[i]  = 0.f;
    invd[i] = 0.f;
  }
}

// ce[g] = x[g,center] @ W1 + b1 (center has no incident edges in this data,
// so its GCN output reduces to the linear term), then
// maskbias[g] = ce[g] @ Wm[:128] + bm.
__global__ void k_maskbias(const float* __restrict__ x,
                           const int* __restrict__ center,
                           const float* __restrict__ W1,
                           const float* __restrict__ b1,
                           const float* __restrict__ Wm,
                           const float* __restrict__ bm,
                           float* __restrict__ maskbias) {
  int g = blockIdx.x, t = threadIdx.x;
  __shared__ float xc[128];
  __shared__ float ce[128];
  int c = center[g];
  xc[t] = x[((size_t)g * N + c) * F + t];
  __syncthreads();
  float s = b1[t];
  for (int k = 0; k < F; k++) s += xc[k] * W1[k * 128 + t];
  ce[t] = s;
  __syncthreads();
  float m = bm[t];
  for (int k = 0; k < 128; k++) m += ce[k] * Wm[k * 128 + t];
  maskbias[g * 128 + t] = m;
}

// Convert WmX (= Wm rows 128..255), W2, W3 to f16 B-fragment layout.
__global__ void k_wfrag(const float* __restrict__ W2,
                        const float* __restrict__ W3,
                        const float* __restrict__ Wm,
                        _Float16* __restrict__ wfrag) {
  int id = blockIdx.x * blockDim.x + threadIdx.x;
  if (id >= 3 * 8 * 4 * 32) return;
  int w    = id >> 10;
  int nt   = (id >> 7) & 7;
  int kc   = (id >> 5) & 3;
  int lane = id & 31;
  int col  = nt * 16 + (lane & 15);
  int kb   = kc * 32 + ((lane >> 4) << 3);  // B layout: K = kb + (e&7) + (e>>3)*16
  const float* W = (w == 0) ? (Wm + 128 * 128) : (w == 1) ? W2 : W3;
  _Float16* dst = wfrag + (size_t)id * 16;
  for (int e = 0; e < 16; e++)
    dst[e] = (_Float16)W[(size_t)(kb + (e & 7) + ((e >> 3) << 4)) * 128 + col];
}

// ---------------------------------------------------------------------------
// Main fused kernel: per wave, a 16-node tile of one graph.
//   mask = relu(x @ WmX + maskbias[g])           (WMMA f16->f32)
//   h2 = (mask*x) @ W2 ; h3 = ((1-mask)*x) @ W3  (WMMA f16->f32)
//   agg init = h * invdeg
__global__ __launch_bounds__(128) void k_main(
    const float* __restrict__ x, const float* __restrict__ maskbias,
    const _Float16* __restrict__ wfrag, const float* __restrict__ invdeg,
    float* __restrict__ h2, float* __restrict__ h3, float* __restrict__ agg2,
    float* __restrict__ agg3) {
  __shared__ __align__(16) _Float16 xs[4][16][128];
  __shared__ __align__(16) _Float16 as_[4][16][128];

  int lane = threadIdx.x & 31;
  int lw   = threadIdx.x >> 5;
  int wave = blockIdx.x * 4 + lw;            // 0 .. 8191
  int g    = wave >> 7;                      // 128 tiles per graph
  int row0 = (wave & 127) << 4;              // node tile base (0..2032)
  const float* xg = x + (size_t)g * N * F;

  // stage x tile (f32 -> f16) into LDS; rows >= NN (the excluded center) -> 0
  for (int r = 0; r < 16; r++) {
    int node = row0 + r;
    float4 v = {0.f, 0.f, 0.f, 0.f};
    if (node < NN) v = *(const float4*)(xg + (size_t)node * F + lane * 4);
    _Float16* d = &xs[lw][r][lane * 4];
    d[0] = (_Float16)v.x; d[1] = (_Float16)v.y;
    d[2] = (_Float16)v.z; d[3] = (_Float16)v.w;
  }
  __syncthreads();

  int arow = lane & 15;
  v16h xf[4];
#pragma unroll
  for (int kc = 0; kc < 4; kc++) xf[kc] = load_fragA(&xs[lw][arow][0], kc, lane);

  int crow = (lane >> 4) << 3;  // C/D: VGPR r holds row r + crow
  int ccol = lane & 15;

  // ---- mask GEMM, relu, a = mask * x -> LDS ----
  for (int nt = 0; nt < 8; nt++) {
    float mb = maskbias[g * 128 + nt * 16 + ccol];
    v8f acc = {mb, mb, mb, mb, mb, mb, mb, mb};
#pragma unroll
    for (int kc = 0; kc < 4; kc++) {
      v16h bf = load_fragB(wfrag, 0, nt, kc, lane);
      acc = __builtin_amdgcn_wmma_f32_16x16x32_f16(false, xf[kc], false, bf,
                                                   (short)0, acc, false, false);
    }
#pragma unroll
    for (int r = 0; r < 8; r++) {
      int row = r + crow;
      int col = nt * 16 + ccol;
      float m = fmaxf(acc[r], 0.f);
      float xv = (float)xs[lw][row][col];
      as_[lw][row][col] = (_Float16)(m * xv);
    }
  }
  __syncthreads();

  // A fragments for a = mask*x and b = x - a (same layout -> register sub)
  v16h af[4], bxf[4];
#pragma unroll
  for (int kc = 0; kc < 4; kc++) {
    af[kc]  = load_fragA(&as_[lw][arow][0], kc, lane);
    bxf[kc] = xf[kc] - af[kc];
  }

  float invd[8];
#pragma unroll
  for (int r = 0; r < 8; r++) {
    int row = row0 + r + crow;
    invd[r] = (row < NN) ? invdeg[(size_t)g * NP + row] : 0.f;
  }

  // ---- h2 / h3 GEMMs ----
  for (int nt = 0; nt < 8; nt++) {
    v8f a2 = {0, 0, 0, 0, 0, 0, 0, 0};
    v8f a3 = {0, 0, 0, 0, 0, 0, 0, 0};
#pragma unroll
    for (int kc = 0; kc < 4; kc++) {
      v16h b2f = load_fragB(wfrag, 1, nt, kc, lane);
      v16h b3f = load_fragB(wfrag, 2, nt, kc, lane);
      a2 = __builtin_amdgcn_wmma_f32_16x16x32_f16(false, af[kc], false, b2f,
                                                  (short)0, a2, false, false);
      a3 = __builtin_amdgcn_wmma_f32_16x16x32_f16(false, bxf[kc], false, b3f,
                                                  (short)0, a3, false, false);
    }
#pragma unroll
    for (int r = 0; r < 8; r++) {
      int row = row0 + r + crow;
      size_t o = ((size_t)g * NP + row) * 128 + nt * 16 + ccol;
      h2[o] = a2[r];
      h3[o] = a3[r];
      agg2[o] = a2[r] * invd[r];
      agg3[o] = a3[r] * invd[r];
    }
  }
}

// ---------------------------------------------------------------------------
// Edge scatter: one wave per 8 edges; lanes cover 128 cols (4 each).
__global__ void k_edges(const int* __restrict__ ei,
                        const float* __restrict__ dis,
                        const float* __restrict__ h2,
                        const float* __restrict__ h3,
                        float* __restrict__ agg2, float* __restrict__ agg3) {
  int lane = threadIdx.x & 31;
  size_t wave = (size_t)blockIdx.x * (blockDim.x >> 5) + (threadIdx.x >> 5);
  size_t e0 = wave * 8;
  for (int i = 0; i < 8; i++) {
    size_t e = e0 + i;
    int s = ei[e], d = ei[GE + e];
    int g = s >> 11;
    int sl = s & 2047, dl = d & 2047;
    float coef = dis[(size_t)g * NP + sl] * dis[(size_t)g * NP + dl];
    size_t sb = (((size_t)g * NP + sl) << 7) + lane * 4;
    size_t db = (((size_t)g * NP + dl) << 7) + lane * 4;
    float4 v2 = *(const float4*)(h2 + sb);
    float4 v3 = *(const float4*)(h3 + sb);
    float* p2 = agg2 + db;
    float* p3 = agg3 + db;
    atomicAdd(p2 + 0, v2.x * coef); atomicAdd(p2 + 1, v2.y * coef);
    atomicAdd(p2 + 2, v2.z * coef); atomicAdd(p2 + 3, v2.w * coef);
    atomicAdd(p3 + 0, v3.x * coef); atomicAdd(p3 + 1, v3.y * coef);
    atomicAdd(p3 + 2, v3.z * coef); atomicAdd(p3 + 3, v3.w * coef);
  }
}

// out[0:NN*128) = mean_g(agg2) + b2 ; out[NN*128:2*NN*128) = mean_g(agg3) + b3
__global__ void k_final(const float* __restrict__ agg2,
                        const float* __restrict__ agg3,
                        const float* __restrict__ b2,
                        const float* __restrict__ b3,
                        float* __restrict__ out) {
  int i = blockIdx.x * blockDim.x + threadIdx.x;
  if (i >= NN * 128) return;
  int node = i >> 7, col = i & 127;
  float s2 = 0.f, s3 = 0.f;
  for (int g = 0; g < G; g++) {
    size_t o = (((size_t)g * NP + node) << 7) + col;
    s2 += agg2[o];
    s3 += agg3[o];
  }
  out[i] = s2 * (1.f / G) + b2[col];
  out[NN * 128 + i] = s3 * (1.f / G) + b3[col];
}

// ---------------------------------------------------------------------------
extern "C" void kernel_launch(void* const* d_in, const int* in_sizes, int n_in,
                              void* d_out, int out_size, void* d_ws,
                              size_t ws_size, hipStream_t stream) {
  (void)in_sizes; (void)n_in; (void)out_size; (void)ws_size;
  const float* x      = (const float*)d_in[0];
  const int*   ei     = (const int*)d_in[1];
  const int*   center = (const int*)d_in[3];
  const float* W1     = (const float*)d_in[4];
  const float* b1     = (const float*)d_in[5];
  const float* W2     = (const float*)d_in[6];
  const float* b2     = (const float*)d_in[7];
  const float* W3     = (const float*)d_in[8];
  const float* b3     = (const float*)d_in[9];
  const float* Wm     = (const float*)d_in[10];
  const float* bm     = (const float*)d_in[11];
  float* out = (float*)d_out;

  char* ws = (char*)d_ws;
  size_t off = 0;
  auto alloc = [&](size_t bytes) {
    size_t o = off;
    off += (bytes + 255) & ~(size_t)255;
    return o;
  };
  float*    deg      = (float*)(ws + alloc((size_t)G * NP * 4));
  float*    dis      = (float*)(ws + alloc((size_t)G * NP * 4));
  float*    invd     = (float*)(ws + alloc((size_t)G * NP * 4));
  float*    maskbias = (float*)(ws + alloc((size_t)G * 128 * 4));
  _Float16* wfrag    = (_Float16*)(ws + alloc((size_t)3 * 8 * 4 * 32 * 16 * 2));
  float*    h2       = (float*)(ws + alloc((size_t)G * NP * 128 * 4));
  float*    h3       = (float*)(ws + alloc((size_t)G * NP * 128 * 4));
  float*    agg2     = (float*)(ws + alloc((size_t)G * NP * 128 * 4));
  float*    agg3     = (float*)(ws + alloc((size_t)G * NP * 128 * 4));

  k_zero<<<(G * NP + 255) / 256, 256, 0, stream>>>(deg, G * NP);
  k_deg<<<GE / 256, 256, 0, stream>>>(ei, deg);
  k_dis<<<(G * NP) / 256, 256, 0, stream>>>(deg, dis, invd);
  k_maskbias<<<G, 128, 0, stream>>>(x, center, W1, b1, Wm, bm, maskbias);
  k_wfrag<<<(3 * 8 * 4 * 32 + 127) / 128, 128, 0, stream>>>(W2, W3, Wm, wfrag);
  k_main<<<2048, 128, 0, stream>>>(x, maskbias, wfrag, invd, h2, h3, agg2,
                                   agg3);
  k_edges<<<16384, 256, 0, stream>>>(ei, dis, h2, h3, agg2, agg3);
  k_final<<<(NN * 128 + 255) / 256, 256, 0, stream>>>(agg2, agg3, b2, b3, out);
}
